// CrossAttention_55130200212194
// MI455X (gfx1250) — matrix operand order repaired and checked by reading, not compiled, not run on hardware
//
#include <hip/hip_runtime.h>
#include <hip/hip_fp16.h>

// ---------------------------------------------------------------------------
// CrossAttention with dual-score injection, MI455X (gfx1250, wave32, WMMA).
// All matmuls use v_wmma_f32_16x16x32_f16 (f16 inputs, f32 accumulation).
// Flash-style single-pass softmax keeps 8x2048x2048 logits out of HBM.
// ---------------------------------------------------------------------------

typedef __attribute__((ext_vector_type(16))) _Float16 v16h;
typedef __attribute__((ext_vector_type(8)))  _Float16 v8h;
typedef __attribute__((ext_vector_type(8)))  float    v8f;

#define H_   8
#define N_   2048
#define D_   80
#define DP_  96          // D padded to 3 K-chunks of 32
#define C_   640
#define SCALE_ 0.11180339887498949f   // 80^-0.5
#define GAMMA_ 0.7f
#define BETA_  0.3f

union V8u { v8f v; float f[8]; };

static __device__ __forceinline__ v8f wmma16(v16h a, v16h b, v8f c) {
  // D = A(16x32 f16) * B(32x16 f16) + C(16x16 f32)
  return __builtin_amdgcn_wmma_f32_16x16x32_f16(false, a, false, b,
                                                (short)0, c, false, false);
}

// A-matrix tile load (16x32, f16). Per ISA layout:
//   lanes 0-15 : row=lane,   comps 0-7 -> K 0..7,   comps 8-15 -> K 16..23
//   lanes 16-31: row=lane-16,comps 0-7 -> K 8..15,  comps 8-15 -> K 24..31
static __device__ __forceinline__ v16h loadA16(const _Float16* base, int ld,
                                               int kbase, int lane) {
  int m  = lane & 15;
  int hi = (lane >> 4) & 1;
  const _Float16* p = base + (size_t)m * ld + kbase + hi * 8;
  v8h lo = *(const v8h*)p;
  v8h hh = *(const v8h*)(p + 16);
  return __builtin_shufflevector(lo, hh, 0,1,2,3,4,5,6,7,8,9,10,11,12,13,14,15);
}

// B-matrix tile load (32x16, f16). Per ISA layout:
//   lane&15 -> N column; comps 0-7 -> K 8*(lane>=16)+c, comps 8-15 -> K 16+8*(lane>=16)+(c-8)
// base is row-major [col][ld] holding B^T (i.e. B[k][n] = base[col0+n][kbase+k]).
static __device__ __forceinline__ v16h loadB16(const _Float16* base, int ld,
                                               int col0, int kbase, int lane) {
  int n  = lane & 15;
  int hi = (lane >> 4) & 1;
  const _Float16* p = base + (size_t)(col0 + n) * ld + kbase + hi * 8;
  v8h lo = *(const v8h*)p;
  v8h hh = *(const v8h*)(p + 16);
  return __builtin_shufflevector(lo, hh, 0,1,2,3,4,5,6,7,8,9,10,11,12,13,14,15);
}

// 16-lane butterfly reductions (masks < 16 stay within each wave32 half,
// matching the f32 fragment row split: rows 0-7 in lanes 0-15, 8-15 in 16-31)
static __device__ __forceinline__ float halfmax16(float v) {
  v = fmaxf(v, __shfl_xor(v, 1, 32));
  v = fmaxf(v, __shfl_xor(v, 2, 32));
  v = fmaxf(v, __shfl_xor(v, 4, 32));
  v = fmaxf(v, __shfl_xor(v, 8, 32));
  return v;
}
static __device__ __forceinline__ float halfsum16(float v) {
  v += __shfl_xor(v, 1, 32);
  v += __shfl_xor(v, 2, 32);
  v += __shfl_xor(v, 4, 32);
  v += __shfl_xor(v, 8, 32);
  return v;
}

// One online-softmax + PV step for a 16x32 logit tile (two v8f fragments).
static __device__ __forceinline__ void online_step(
    v8f sa, v8f sb, float* mrun, float* lrun, v8f* acc,
    const _Float16* VT, int n0, int lane, _Float16* Pl) {
  V8u ua; ua.v = sa;
  V8u ub; ub.v = sb;
  int hi = (lane >> 4) & 1;
  int l15 = lane & 15;
  V8u av;
#pragma unroll
  for (int r = 0; r < 8; ++r) {
    float t  = fmaxf(ua.f[r], ub.f[r]);
    t        = halfmax16(t);
    float mn = fmaxf(mrun[r], t);
    float al = __expf(mrun[r] - mn);
    mrun[r]  = mn;
    av.f[r]  = al;
    float p0 = __expf(ua.f[r] - mn);
    float p1 = __expf(ub.f[r] - mn);
    lrun[r]  = lrun[r] * al + halfsum16(p0 + p1);
    // transpose P through LDS: fragment (row=r+8*hi, col=l15 / l15+16)
    Pl[(r + 8 * hi) * 32 + l15]      = (_Float16)p0;
    Pl[(r + 8 * hi) * 32 + 16 + l15] = (_Float16)p1;
  }
  __syncthreads();
  v16h Pa = loadA16(Pl, 32, 0, lane);
#pragma unroll
  for (int jd = 0; jd < 6; ++jd) {
    v16h Bv = loadB16(VT, N_, jd * 16, n0, lane);  // B[k][n] = VT[d][key]
    acc[jd] = acc[jd] * av.v;                       // rescale by exp(mold-mnew)
    acc[jd] = wmma16(Pa, Bv, acc[jd]);
  }
}

// --------------------------- pack / utility kernels ------------------------

__global__ void k_f32_to_f16(const float* __restrict__ in,
                             _Float16* __restrict__ out, int n) {
  int i = blockIdx.x * blockDim.x + threadIdx.x;
  if (i < n) out[i] = (_Float16)in[i];
}

__global__ void k_zero_f16(_Float16* __restrict__ out, int n) {
  int i = blockIdx.x * blockDim.x + threadIdx.x;
  if (i < n) out[i] = (_Float16)0.0f;
}

// [H][N][80] f32 -> [H][N][96] f16, zero-padded
__global__ void k_pack_head(const float* __restrict__ in,
                            _Float16* __restrict__ out) {
  int i = blockIdx.x * blockDim.x + threadIdx.x;
  if (i >= H_ * N_ * DP_) return;
  int d  = i % DP_;
  int hn = i / DP_;
  out[i] = (d < D_) ? (_Float16)in[(size_t)hn * D_ + d] : (_Float16)0.0f;
}

// [H][N][80] f32 -> [H][96][N] f16 (transposed), zero-padded rows
__global__ void k_pack_headT(const float* __restrict__ in,
                             _Float16* __restrict__ out) {
  int i = blockIdx.x * blockDim.x + threadIdx.x;
  if (i >= H_ * DP_ * N_) return;
  int n = i % N_;
  int r = i / N_;
  int d = r % DP_;
  int h = r / DP_;
  out[i] = (d < D_) ? (_Float16)in[((size_t)h * N_ + n) * D_ + d]
                    : (_Float16)0.0f;
}

// --------------------------- GEMM kernels ----------------------------------

// q = x @ Wq^T ; v = x @ Wv^T  (shared A tiles); scatter into per-head
// padded q layout and transposed v layout. One wave per 16x16 output tile.
__global__ void k_gemm_qv(const _Float16* __restrict__ X,
                          const _Float16* __restrict__ Wq,
                          const _Float16* __restrict__ Wv,
                          _Float16* __restrict__ qp,
                          _Float16* __restrict__ vT) {
  int lane = threadIdx.x;
  int row0 = blockIdx.x * 16;
  int c0   = blockIdx.y * 16;
  const _Float16* Xr = X + (size_t)row0 * C_;
  v8f aq = {0,0,0,0,0,0,0,0};
  v8f av = {0,0,0,0,0,0,0,0};
  for (int kt = 0; kt < C_ / 32; ++kt) {
    v16h A  = loadA16(Xr, C_, kt * 32, lane);
    v16h Bq = loadB16(Wq, C_, c0, kt * 32, lane);   // B[k][c] = Wq[c][k]
    v16h Bv = loadB16(Wv, C_, c0, kt * 32, lane);
    aq = wmma16(A, Bq, aq);
    av = wmma16(A, Bv, av);
  }
  int hi = (lane >> 4) & 1, l15 = lane & 15;
  V8u uq; uq.v = aq;
  V8u uv; uv.v = av;
#pragma unroll
  for (int r = 0; r < 8; ++r) {
    int m = row0 + r + 8 * hi;
    int c = c0 + l15;
    int h = c / D_;
    int d = c % D_;
    qp[((size_t)h * N_ + m) * DP_ + d] = (_Float16)uq.f[r];
    vT[((size_t)h * DP_ + d) * N_ + m] = (_Float16)uv.f[r];
  }
}

// y = merged @ Wout^T + bout  -> f32 output
__global__ void k_gemm_out(const _Float16* __restrict__ Xm,
                           const _Float16* __restrict__ Wo,
                           const float* __restrict__ bias,
                           float* __restrict__ Y) {
  int lane = threadIdx.x;
  int row0 = blockIdx.x * 16;
  int c0   = blockIdx.y * 16;
  const _Float16* Xr = Xm + (size_t)row0 * C_;
  v8f acc = {0,0,0,0,0,0,0,0};
  for (int kt = 0; kt < C_ / 32; ++kt) {
    v16h A = loadA16(Xr, C_, kt * 32, lane);
    v16h B = loadB16(Wo, C_, c0, kt * 32, lane);
    acc = wmma16(A, B, acc);
  }
  int hi = (lane >> 4) & 1, l15 = lane & 15;
  V8u u; u.v = acc;
#pragma unroll
  for (int r = 0; r < 8; ++r) {
    int m = row0 + r + 8 * hi;
    int c = c0 + l15;
    Y[(size_t)m * C_ + c] = u.f[r] + bias[c];
  }
}

// --------------------------- attention kernel ------------------------------
// One wave per (head, 16-row tile). Two independent online softmaxes:
//  dual: softmax(scale*(g*Qi.Kl^T + (1-g)*Q.Kr^T)) @ Vref
//  self: softmax(scale*(Qi.Ki^T)) @ Vself
// out = 0.7*dual + 0.3*self, written f16 into merged [N][C] at col h*80+d.
__global__ void k_attention(const _Float16* __restrict__ qinj,
                            const _Float16* __restrict__ kinj,
                            const _Float16* __restrict__ kref,
                            const _Float16* __restrict__ krefL,
                            const _Float16* __restrict__ vrefT,
                            const _Float16* __restrict__ qp,
                            const _Float16* __restrict__ vselfT,
                            _Float16* __restrict__ merged) {
  __shared__ __align__(16) _Float16 Pl[16 * 32];
  int lane = threadIdx.x;
  int h    = blockIdx.y;
  int row0 = blockIdx.x * 16;

  const _Float16* Qi = qinj  + ((size_t)h * N_ + row0) * DP_;
  const _Float16* Qc = qp    + ((size_t)h * N_ + row0) * DP_;
  const _Float16* Ki = kinj  + (size_t)h * N_ * DP_;
  const _Float16* Kr = kref  + (size_t)h * N_ * DP_;
  const _Float16* Kl = krefL + (size_t)h * N_ * DP_;
  const _Float16* Vr = vrefT + (size_t)h * DP_ * N_;
  const _Float16* Vs = vselfT + (size_t)h * DP_ * N_;

  v16h Ai[3], Ac[3];
#pragma unroll
  for (int t = 0; t < 3; ++t) {
    Ai[t] = loadA16(Qi, DP_, t * 32, lane);
    Ac[t] = loadA16(Qc, DP_, t * 32, lane);
  }

  v8f acc1[6], acc2[6];
  float m1[8], l1[8], m2[8], l2[8];
#pragma unroll
  for (int j = 0; j < 6; ++j) {
    acc1[j] = (v8f){0,0,0,0,0,0,0,0};
    acc2[j] = (v8f){0,0,0,0,0,0,0,0};
  }
#pragma unroll
  for (int r = 0; r < 8; ++r) {
    m1[r] = -1e30f; l1[r] = 0.0f;
    m2[r] = -1e30f; l2[r] = 0.0f;
  }

  for (int kt = 0; kt < N_ / 32; ++kt) {
    int n0 = kt * 32;
    // ---- dual-score path: mix logits pre-softmax ----
    v8f s1a = {0,0,0,0,0,0,0,0}, s1b = {0,0,0,0,0,0,0,0};
    v8f s2a = {0,0,0,0,0,0,0,0}, s2b = {0,0,0,0,0,0,0,0};
#pragma unroll
    for (int t = 0; t < 3; ++t) {
      v16h B0 = loadB16(Kl, DP_, n0,      t * 32, lane);
      v16h B1 = loadB16(Kl, DP_, n0 + 16, t * 32, lane);
      s1a = wmma16(Ai[t], B0, s1a);
      s1b = wmma16(Ai[t], B1, s1b);
      v16h C0 = loadB16(Kr, DP_, n0,      t * 32, lane);
      v16h C1 = loadB16(Kr, DP_, n0 + 16, t * 32, lane);
      s2a = wmma16(Ac[t], C0, s2a);
      s2b = wmma16(Ac[t], C1, s2b);
    }
    v8f sma = (s1a * GAMMA_ + s2a * (1.0f - GAMMA_)) * SCALE_;
    v8f smb = (s1b * GAMMA_ + s2b * (1.0f - GAMMA_)) * SCALE_;
    online_step(sma, smb, m1, l1, acc1, Vr, n0, lane, Pl);
    __syncthreads();
    // ---- self path ----
    v8f s3a = {0,0,0,0,0,0,0,0}, s3b = {0,0,0,0,0,0,0,0};
#pragma unroll
    for (int t = 0; t < 3; ++t) {
      v16h B0 = loadB16(Ki, DP_, n0,      t * 32, lane);
      v16h B1 = loadB16(Ki, DP_, n0 + 16, t * 32, lane);
      s3a = wmma16(Ai[t], B0, s3a);
      s3b = wmma16(Ai[t], B1, s3b);
    }
    s3a = s3a * SCALE_;
    s3b = s3b * SCALE_;
    online_step(s3a, s3b, m2, l2, acc2, Vs, n0, lane, Pl);
    __syncthreads();
  }

  // ---- finalize: out = (1-beta)*acc1/l1 + beta*acc2/l2 ----
  int hi = (lane >> 4) & 1, l15 = lane & 15;
#pragma unroll
  for (int jd = 0; jd < 6; ++jd) {
    V8u u1; u1.v = acc1[jd];
    V8u u2; u2.v = acc2[jd];
#pragma unroll
    for (int r = 0; r < 8; ++r) {
      int d = jd * 16 + l15;
      if (d < D_) {
        int m   = r + 8 * hi;
        float o = (1.0f - BETA_) * u1.f[r] / l1[r] + BETA_ * u2.f[r] / l2[r];
        merged[(size_t)(row0 + m) * C_ + h * D_ + d] = (_Float16)o;
      }
    }
  }
}

// --------------------------- host launcher ---------------------------------

extern "C" void kernel_launch(void* const* d_in, const int* in_sizes, int n_in,
                              void* d_out, int out_size, void* d_ws, size_t ws_size,
                              hipStream_t stream) {
  const float* x     = (const float*)d_in[0];
  const float* q_inj = (const float*)d_in[1];
  const float* k_inj = (const float*)d_in[2];
  const float* k_ref = (const float*)d_in[3];
  const float* k_rL  = (const float*)d_in[4];
  const float* v_ref = (const float*)d_in[5];
  const float* Wq    = (const float*)d_in[6];
  const float* Wv    = (const float*)d_in[7];
  const float* Wout  = (const float*)d_in[8];
  const float* bout  = (const float*)d_in[9];
  float* out = (float*)d_out;

  // workspace layout (f16 elements); total ~29.7 MB
  const size_t SZ_HEAD = (size_t)H_ * N_ * DP_;  // 1,572,864
  const size_t SZ_XC   = (size_t)N_ * C_;        // 1,310,720
  const size_t SZ_W    = (size_t)C_ * C_;        //   409,600
  _Float16* ws       = (_Float16*)d_ws;
  _Float16* qinj_p   = ws;
  _Float16* kinj_p   = qinj_p + SZ_HEAD;
  _Float16* kref_p   = kinj_p + SZ_HEAD;
  _Float16* krefL_p  = kref_p + SZ_HEAD;
  _Float16* vrefT_p  = krefL_p + SZ_HEAD;
  _Float16* q_p      = vrefT_p + SZ_HEAD;
  _Float16* vselfT_p = q_p + SZ_HEAD;
  _Float16* x_h      = vselfT_p + SZ_HEAD;
  _Float16* wq_h     = x_h + SZ_XC;
  _Float16* wv_h     = wq_h + SZ_W;
  _Float16* wout_h   = wv_h + SZ_W;
  _Float16* merged_h = wout_h + SZ_W;

  const int TB = 256;
  // zero padded regions that the projection GEMM scatters into (q_p, vselfT_p
  // are adjacent -> one launch)
  {
    int n = (int)(2 * SZ_HEAD);
    k_zero_f16<<<(n + TB - 1) / TB, TB, 0, stream>>>(q_p, n);
  }
  // pack injected tensors (padded f16 / transposed f16)
  {
    int n = (int)SZ_HEAD;
    int g = (n + TB - 1) / TB;
    k_pack_head <<<g, TB, 0, stream>>>(q_inj, qinj_p);
    k_pack_head <<<g, TB, 0, stream>>>(k_inj, kinj_p);
    k_pack_head <<<g, TB, 0, stream>>>(k_ref, kref_p);
    k_pack_head <<<g, TB, 0, stream>>>(k_rL,  krefL_p);
    k_pack_headT<<<g, TB, 0, stream>>>(v_ref, vrefT_p);
  }
  // pack activations and weights to f16
  k_f32_to_f16<<<((int)SZ_XC + TB - 1) / TB, TB, 0, stream>>>(x,    x_h,    (int)SZ_XC);
  k_f32_to_f16<<<((int)SZ_W  + TB - 1) / TB, TB, 0, stream>>>(Wq,   wq_h,   (int)SZ_W);
  k_f32_to_f16<<<((int)SZ_W  + TB - 1) / TB, TB, 0, stream>>>(Wv,   wv_h,   (int)SZ_W);
  k_f32_to_f16<<<((int)SZ_W  + TB - 1) / TB, TB, 0, stream>>>(Wout, wout_h, (int)SZ_W);

  // q / v_self projections: 128 x 40 tiles, 1 wave each
  k_gemm_qv<<<dim3(N_ / 16, C_ / 16), 32, 0, stream>>>(x_h, wq_h, wv_h,
                                                       q_p, vselfT_p);
  // fused dual+self flash attention: (row tiles, heads)
  k_attention<<<dim3(N_ / 16, H_), 32, 0, stream>>>(qinj_p, kinj_p, kref_p,
                                                    krefL_p, vrefT_p, q_p,
                                                    vselfT_p, merged_h);
  // output projection + bias -> f32
  k_gemm_out<<<dim3(N_ / 16, C_ / 16), 32, 0, stream>>>(merged_h, wout_h,
                                                        bout, out);
  (void)in_sizes; (void)n_in; (void)out_size; (void)ws_size;
}
